// SparseBasicBlock_88776974008701
// MI455X (gfx1250) — matrix operand order repaired and checked by reading, not compiled, not run on hardware
//
#include <hip/hip_runtime.h>

// ---------------------------------------------------------------------------
// SparseBasicBlock for MI455X (gfx1250, wave32, WMMA bf16 path)
//   conv: weights resident in LDS, A operands gathered straight from global
//   (L2-resident feature table) in WMMA register layout, zero-sentinel row,
//   ping-pong double-buffered k+1 prefetch (no register copies),
//   32-bit prescaled gather offsets, M=32 rows per wave.
// ---------------------------------------------------------------------------

typedef __attribute__((ext_vector_type(16))) __bf16 v16bf;
typedef __attribute__((ext_vector_type(8)))  float  v8f;
typedef __attribute__((ext_vector_type(8)))  int    v8i_t;

#define KOFF 27
#define CCH  64
#define R_BLOCKS 240

__device__ __forceinline__ unsigned short f2bf(float f) {
  __bf16 h = (__bf16)f;
  return __builtin_bit_cast(unsigned short, h);
}

// ---------------------------------------------------------------------------
// 1) Repack weights [27][64][64] fp32 -> bf16 WMMA-B tiles.
//    Tile (k,c,n): 1KB = two 512B halves; lane's 16 bf16 split as
//    half0[lane*8 + e] (e=0..7), half1[lane*8 + e-8] (e=8..15) so each
//    ds_load_b128 in the conv is bank-conflict-free.
//    Element e of lane l: K = c*32 + (e<8 ? (l/16)*8 + e : 16 + (l/16)*8 + e-8) ; N = n*16 + l%16.
// ---------------------------------------------------------------------------
__global__ void repack_weights_kernel(const float* __restrict__ W1,
                                      const float* __restrict__ W2,
                                      unsigned short* __restrict__ W1b,
                                      unsigned short* __restrict__ W2b) {
  int b = blockIdx.x;           // 0..431 : 216 tiles per weight set
  int lane = threadIdx.x;       // 0..31
  const float* Ws = (b < 216) ? W1 : W2;
  unsigned short* Wd = (b < 216) ? W1b : W2b;
  int t = b % 216;
  int k = t >> 3;               // kernel offset
  int c = (t >> 2) & 1;         // K chunk (rows 0-31 / 32-63)
  int n = t & 3;                // N tile (cols 16n..16n+15)
  int hi = lane >> 4;
  int nc = n * 16 + (lane & 15);
  unsigned short* dst = Wd + ((k * 2 + c) * 4 + n) * 512;
  const float* src = Ws + k * (CCH * CCH);
#pragma unroll
  for (int e = 0; e < 16; ++e) {
    int K = c * 32 + (e < 8 ? hi * 8 + e : 16 + hi * 8 + (e - 8));
    unsigned short v = f2bf(src[K * CCH + nc]);
    if (e < 8) dst[lane * 8 + e] = v;
    else       dst[256 + lane * 8 + (e - 8)] = v;
  }
}

// ---------------------------------------------------------------------------
// 2) fp32 -> bf16 conversion, padded with one zeroed sentinel row (row N)
// ---------------------------------------------------------------------------
__global__ void cvt_bf16_kernel(const float* __restrict__ x,
                                unsigned short* __restrict__ o,
                                int total, int total_pad) {
  int i = blockIdx.x * 256 + threadIdx.x;
  if (i >= total_pad) return;
  o[i] = (i < total) ? f2bf(x[i]) : (unsigned short)0;
}

// ---------------------------------------------------------------------------
// helpers
// ---------------------------------------------------------------------------
__device__ __forceinline__ v16bf load_btile(const unsigned short* tp, int lane) {
  uint4 lo = *(const uint4*)(tp + lane * 8);
  uint4 hi = *(const uint4*)(tp + 256 + lane * 8);
  v8i_t v = { (int)lo.x, (int)lo.y, (int)lo.z, (int)lo.w,
              (int)hi.x, (int)hi.y, (int)hi.z, (int)hi.w };
  return __builtin_bit_cast(v16bf, v);
}

__device__ __forceinline__ v16bf packA(const uint4& a, const uint4& b) {
  v8i_t v = { (int)a.x, (int)a.y, (int)a.z, (int)a.w,
              (int)b.x, (int)b.y, (int)b.z, (int)b.w };
  return __builtin_bit_cast(v16bf, v);
}

// Gather both row-tiles' A fragments for offset k into buf.
// idx table holds prescaled byte offsets (src*128); +hi*16 selects lane half.
__device__ __forceinline__ void gatherK(const unsigned short* __restrict__ Fbf,
                                        const int* __restrict__ idxw,
                                        int lmod, int hi16, int k,
                                        uint4 (&buf)[2][4]) {
#pragma unroll
  for (int t = 0; t < 2; ++t) {
    int off = idxw[(t * 16 + lmod) * KOFF + k] + hi16;
    const char* rp = (const char*)Fbf + off;
    buf[t][0] = *(const uint4*)(rp);
    buf[t][1] = *(const uint4*)(rp + 32);
    buf[t][2] = *(const uint4*)(rp + 64);
    buf[t][3] = *(const uint4*)(rp + 96);
  }
}

// 16 WMMAs for one kernel offset: 4 N-tiles x 2 K-chunks x 2 row-tiles.
__device__ __forceinline__ void computeK(const unsigned short* __restrict__ wk,
                                         const uint4 (&buf)[2][4],
                                         v8f (&acc)[2][4], int lane) {
  v16bf a00 = packA(buf[0][0], buf[0][1]);
  v16bf a01 = packA(buf[0][2], buf[0][3]);
  v16bf a10 = packA(buf[1][0], buf[1][1]);
  v16bf a11 = packA(buf[1][2], buf[1][3]);
#pragma unroll
  for (int n = 0; n < 4; ++n) {
    v16bf b0 = load_btile(wk + n * 512, lane);
    v16bf b1 = load_btile(wk + (4 + n) * 512, lane);
    acc[0][n] = __builtin_amdgcn_wmma_f32_16x16x32_bf16(
        false, a00, false, b0, (short)0, acc[0][n], false, false);
    acc[0][n] = __builtin_amdgcn_wmma_f32_16x16x32_bf16(
        false, a01, false, b1, (short)0, acc[0][n], false, false);
    acc[1][n] = __builtin_amdgcn_wmma_f32_16x16x32_bf16(
        false, a10, false, b0, (short)0, acc[1][n], false, false);
    acc[1][n] = __builtin_amdgcn_wmma_f32_16x16x32_bf16(
        false, a11, false, b1, (short)0, acc[1][n], false, false);
  }
}

// ---------------------------------------------------------------------------
// 3) Gather-GEMM conv:  Y[i,:] = sum_k mask(i,k) * Fbf[nidx(i,k),:] @ W[k]
//    256 threads = 8 waves; wave owns 32 voxels (2 A tiles) x 64 out chans.
// ---------------------------------------------------------------------------
__global__ __launch_bounds__(256) void subm_conv_kernel(
    const unsigned short* __restrict__ Fbf,   // [N+1][64] bf16 (row N zeros)
    const unsigned short* __restrict__ Wb,    // repacked [27][2][4][512] bf16
    const int* __restrict__ nidx,             // [N][27]
    const unsigned char* __restrict__ nmask,  // [N][27] (bool)
    float* __restrict__ Y,                    // [N][64] fp32 out
    int N) {
  __shared__ __align__(16) unsigned short Wlds[KOFF * 4096];  // 221184 B
  __shared__ int idxbuf[8][32 * KOFF];                        // 27648 B

  const int tid = threadIdx.x;
  const int wave = tid >> 5;
  const int lane = tid & 31;
  const int hi = lane >> 4;
  const int hi16 = hi * 16;
  const int lmod = lane & 15;
  const int rowBase = blockIdx.x * 256 + wave * 32;

  // ---- cooperative: whole repacked weight set -> LDS ----
  {
    const uint4* src = (const uint4*)Wb;
    uint4* dst = (uint4*)Wlds;
    for (int i = tid; i < KOFF * 512; i += 256) dst[i] = src[i];
  }
  // ---- per-wave: fold idx+mask, store prescaled byte offsets (src*128) ----
  int* idxw = idxbuf[wave];
  for (int j = lane; j < 32 * KOFF; j += 32) {
    int r = j / KOFF;
    int kk = j - r * KOFF;
    int grow = rowBase + r;
    int v = N;  // sentinel: zeroed row
    if (grow < N) {
      if (nmask[grow * KOFF + kk]) v = nidx[grow * KOFF + kk];
    }
    idxw[j] = v << 7;   // byte offset of row
  }
  __syncthreads();   // Wlds visible to all waves; idx table is wave-private

  v8f acc[2][4];
#pragma unroll
  for (int t = 0; t < 2; ++t)
#pragma unroll
    for (int n = 0; n < 4; ++n)
      acc[t][n] = (v8f){0.f,0.f,0.f,0.f,0.f,0.f,0.f,0.f};

  // ---- ping-pong double-buffered pipeline over 27 offsets ----
  uint4 bufA[2][4], bufB[2][4];
  gatherK(Fbf, idxw, lmod, hi16, 0, bufA);

#pragma unroll 1
  for (int k = 0; k < KOFF; k += 2) {
    if (k + 1 < KOFF) gatherK(Fbf, idxw, lmod, hi16, k + 1, bufB);
    computeK(Wlds + k * 4096, bufA, acc, lane);
    if (k + 1 < KOFF) {
      if (k + 2 < KOFF) gatherK(Fbf, idxw, lmod, hi16, k + 2, bufA);
      computeK(Wlds + (k + 1) * 4096, bufB, acc, lane);
    }
  }

  // ---- write D: lane l, VGPR v -> (row = t*16 + v + 8*hi, col = n*16+l%16) --
#pragma unroll
  for (int t = 0; t < 2; ++t) {
    if (rowBase + t * 16 >= N) break;
#pragma unroll
    for (int n = 0; n < 4; ++n) {
#pragma unroll
      for (int v = 0; v < 8; ++v) {
        int row = rowBase + t * 16 + v + 8 * hi;
        if (row < N) Y[(size_t)row * CCH + n * 16 + lmod] = acc[t][n][v];
      }
    }
  }
}

// ---------------------------------------------------------------------------
// 4) Deterministic per-channel sum / sum-of-squares block partials
// ---------------------------------------------------------------------------
__global__ __launch_bounds__(256) void reduce_kernel(const float* __restrict__ Y,
                                                     float* __restrict__ psum,
                                                     float* __restrict__ psq,
                                                     int N) {
  __shared__ float ls[256], ls2[256];
  int t = threadIdx.x;
  int c = t & 63;
  int rsub = t >> 6;
  int stride = gridDim.x * 4;
  float s = 0.f, s2 = 0.f;
  for (int r = blockIdx.x * 4 + rsub; r < N; r += stride) {
    float v = Y[(size_t)r * CCH + c];
    s += v; s2 += v * v;
  }
  ls[t] = s; ls2[t] = s2;
  __syncthreads();
  if (t < 64) {
    float a = ls[t] + ls[t + 64] + ls[t + 128] + ls[t + 192];
    float b = ls2[t] + ls2[t + 64] + ls2[t + 128] + ls2[t + 192];
    psum[blockIdx.x * 64 + t] = a;
    psq[blockIdx.x * 64 + t] = b;
  }
}

// ---------------------------------------------------------------------------
// 5) Fold partials -> per-channel BN scale/bias (scale = g*rsqrt(var+eps))
// ---------------------------------------------------------------------------
__global__ void stats_kernel(const float* __restrict__ psum,
                             const float* __restrict__ psq,
                             const float* __restrict__ gamma,
                             const float* __restrict__ beta,
                             float* __restrict__ scale,
                             float* __restrict__ bias,
                             int N, int nblk) {
  int c = threadIdx.x;
  float s = 0.f, s2 = 0.f;
  for (int b = 0; b < nblk; ++b) { s += psum[b * 64 + c]; s2 += psq[b * 64 + c]; }
  float inv = 1.0f / (float)N;
  float mu = s * inv;
  float var = s2 * inv - mu * mu;
  float sc = gamma[c] * rsqrtf(var + 1e-3f);
  scale[c] = sc;
  bias[c] = beta[c] - mu * sc;
}

// ---------------------------------------------------------------------------
// 6) BN1 + ReLU, emitting bf16 conv2 input (sentinel row kept zeroed)
// ---------------------------------------------------------------------------
__global__ void bnrelu_cvt_kernel(const float* __restrict__ Y,
                                  const float* __restrict__ scale,
                                  const float* __restrict__ bias,
                                  unsigned short* __restrict__ X,
                                  int total, int total_pad) {
  int i = blockIdx.x * 256 + threadIdx.x;
  if (i >= total_pad) return;
  if (i < total) {
    int c = i & 63;
    float v = fmaxf(fmaf(Y[i], scale[c], bias[c]), 0.0f);
    X[i] = f2bf(v);
  } else {
    X[i] = 0;
  }
}

// ---------------------------------------------------------------------------
// 7) BN2 + residual + ReLU (in place on d_out)
// ---------------------------------------------------------------------------
__global__ void final_kernel(float* __restrict__ out,
                             const float* __restrict__ feat,
                             const float* __restrict__ scale,
                             const float* __restrict__ bias, int total) {
  int i = blockIdx.x * 256 + threadIdx.x;
  if (i >= total) return;
  int c = i & 63;
  out[i] = fmaxf(fmaf(out[i], scale[c], bias[c]) + feat[i], 0.0f);
}

// ---------------------------------------------------------------------------
// Host-side orchestration
// ---------------------------------------------------------------------------
static inline size_t align_up(size_t x, size_t a) { return (x + a - 1) & ~(a - 1); }

extern "C" void kernel_launch(void* const* d_in, const int* in_sizes, int n_in,
                              void* d_out, int out_size, void* d_ws, size_t ws_size,
                              hipStream_t stream) {
  const float* features = (const float*)d_in[0];
  const float* W1 = (const float*)d_in[1];
  const float* gamma1 = (const float*)d_in[2];
  const float* beta1 = (const float*)d_in[3];
  const float* W2 = (const float*)d_in[4];
  const float* gamma2 = (const float*)d_in[5];
  const float* beta2 = (const float*)d_in[6];
  const int* nidx = (const int*)d_in[7];
  const unsigned char* nmask = (const unsigned char*)d_in[8];

  const int N = in_sizes[0] / CCH;
  const int total = N * CCH;
  const int total_pad = total + CCH;   // + zeroed sentinel row

  // workspace carve-up
  char* ws = (char*)d_ws;
  size_t off = 0;
  unsigned short* W1b = (unsigned short*)(ws + off); off += (size_t)KOFF * 4096 * 2;
  unsigned short* W2b = (unsigned short*)(ws + off); off += (size_t)KOFF * 4096 * 2;
  off = align_up(off, 256);
  float* scale1 = (float*)(ws + off); off += 64 * 4;
  float* bias1  = (float*)(ws + off); off += 64 * 4;
  float* scale2 = (float*)(ws + off); off += 64 * 4;
  float* bias2  = (float*)(ws + off); off += 64 * 4;
  off = align_up(off, 256);
  float* psum = (float*)(ws + off); off += (size_t)R_BLOCKS * 64 * 4;
  float* psq  = (float*)(ws + off); off += (size_t)R_BLOCKS * 64 * 4;
  off = align_up(off, 512);
  unsigned short* Fbf = (unsigned short*)(ws + off); off += (size_t)total_pad * 2;
  off = align_up(off, 512);
  float* Y1 = (float*)(ws + off); off += (size_t)total * 4;

  float* out = (float*)d_out;

  const int ew_grid = (total + 255) / 256;
  const int ewp_grid = (total_pad + 255) / 256;
  const int conv_grid = (N + 255) / 256;

  // 1) repack both weight tensors into WMMA-B swizzled bf16
  repack_weights_kernel<<<432, 32, 0, stream>>>(W1, W2, W1b, W2b);
  // 2) features -> bf16 (+ zero sentinel row)
  cvt_bf16_kernel<<<ewp_grid, 256, 0, stream>>>(features, Fbf, total, total_pad);
  // 3) conv1
  subm_conv_kernel<<<conv_grid, 256, 0, stream>>>(Fbf, W1b, nidx, nmask, Y1, N);
  // 4) BN1 stats
  reduce_kernel<<<R_BLOCKS, 256, 0, stream>>>(Y1, psum, psq, N);
  stats_kernel<<<1, 64, 0, stream>>>(psum, psq, gamma1, beta1, scale1, bias1, N, R_BLOCKS);
  // 5) BN1 + ReLU -> bf16 conv2 input (reuse Fbf, keep sentinel zeroed)
  bnrelu_cvt_kernel<<<ewp_grid, 256, 0, stream>>>(Y1, scale1, bias1, Fbf, total, total_pad);
  // 6) conv2 -> d_out (pre-BN)
  subm_conv_kernel<<<conv_grid, 256, 0, stream>>>(Fbf, W2b, nidx, nmask, out, N);
  // 7) BN2 stats
  reduce_kernel<<<R_BLOCKS, 256, 0, stream>>>(out, psum, psq, N);
  stats_kernel<<<1, 64, 0, stream>>>(psum, psq, gamma2, beta2, scale2, bias2, N, R_BLOCKS);
  // 8) BN2 + residual + ReLU in place
  final_kernel<<<ew_grid, 256, 0, stream>>>(out, features, scale2, bias2, total);
}
